// MaskedMessagePassingLayer_21895743275106
// MI455X (gfx1250) — matrix operand order repaired and checked by reading, not compiled, not run on hardware
//
#include <hip/hip_runtime.h>
#include <math.h>

// Problem constants (from reference)
#define Bn  32
#define Nn  64
#define Dn  256
#define En  32
#define DNn 64
#define Hn  8
#define Mn  32
#define HM  256   // H*M
#define EPSf 1e-5f

typedef __attribute__((ext_vector_type(16))) __bf16   v16bf;
typedef __attribute__((ext_vector_type(8)))  unsigned v8u;
typedef __attribute__((ext_vector_type(8)))  float    v8f;

__device__ __forceinline__ unsigned fbits(float x) {
    return __builtin_bit_cast(unsigned, x);
}
__device__ __forceinline__ float bits_f(unsigned u) {
    return __builtin_bit_cast(float, u);
}
// Pack truncated-bf16(x1):truncated-bf16(x0) into one dword with v_perm_b32.
__device__ __forceinline__ unsigned bfpack(float x1, float x0) {
    return __builtin_amdgcn_perm(fbits(x1), fbits(x0), 0x07060302u);
}

// ---------------------------------------------------------------------------
// Kernel 1: fused LN(embeddings)*mask -> projections (pre-scaled by 1/sqrt2),
// then exponential factor precompute so the hot kernel's gate is mul-only:
//   rr = { exp(2*recv'), exp(-rn') }   (per receiver j)
//   ss = { exp(2*send'), exp(-sn') }   (per sender i)
// tanh(recv'+send') = (u-1)/(u+1), u = rr.x*ss.x
// sigmoid(rn'+sn')  = 1/(1+f),     f = rr.y*ss.y
// One block per (b,n), 256 threads.
// ---------------------------------------------------------------------------
__global__ void __launch_bounds__(256) proj_kernel(
    const float* __restrict__ embeddings,
    const unsigned char* __restrict__ mask,
    const float* __restrict__ node_features,
    const float* __restrict__ W_recv, const float* __restrict__ W_send,
    const float* __restrict__ W_recv_node, const float* __restrict__ W_send_node,
    float2* __restrict__ rr_ws,
    float2* __restrict__ ss_ws)
{
    const int bn  = blockIdx.x;
    const int tid = threadIdx.x;
    __shared__ float er[Dn];
    __shared__ float nr[DNn];
    __shared__ float ps[8], ps2[8];

    float x = embeddings[(size_t)bn * Dn + tid];
    float s = x, s2 = x * x;
    for (int off = 16; off; off >>= 1) {
        s  += __shfl_xor(s,  off);
        s2 += __shfl_xor(s2, off);
    }
    if ((tid & 31) == 0) { ps[tid >> 5] = s; ps2[tid >> 5] = s2; }
    if (tid < DNn) nr[tid] = node_features[(size_t)bn * DNn + tid];
    __syncthreads();
    float ts = 0.f, ts2 = 0.f;
    #pragma unroll
    for (int k = 0; k < 8; ++k) { ts += ps[k]; ts2 += ps2[k]; }
    float mu  = ts * (1.0f / 256.0f);
    float var = ts2 * (1.0f / 256.0f) - mu * mu;
    float mf  = mask[bn] ? 1.0f : 0.0f;
    er[tid] = (x - mu) * rsqrtf(var + EPSf) * mf;
    __syncthreads();

    float aR = 0.f, aS = 0.f;
    for (int d = 0; d < Dn; ++d) {
        float e = er[d];
        aR = fmaf(e, W_recv[(size_t)d * HM + tid], aR);
        aS = fmaf(e, W_send[(size_t)d * HM + tid], aS);
    }
    float aRN = 0.f, aSN = 0.f;
    for (int d = 0; d < DNn; ++d) {
        float nv = nr[d];
        aRN = fmaf(nv, W_recv_node[(size_t)d * HM + tid], aRN);
        aSN = fmaf(nv, W_send_node[(size_t)d * HM + tid], aSN);
    }
    const float is2 = 0.70710678118654752f;
    size_t o = (size_t)bn * HM + tid;
    // Factorized exponentials (all arguments O(few): safe range for N(0,1)
    // scale inputs; downstream fmin clamp handles any overflow tail).
    rr_ws[o] = make_float2(__expf(2.0f * aR * is2), __expf(-aRN * is2));
    ss_ws[o] = make_float2(__expf(2.0f * aS * is2), __expf(-aSN * is2));
}

// ---------------------------------------------------------------------------
// Kernel 2: main fused message kernel. One block per (b, receiver j).
// Per wave: 2 hm-tiles of 16 cols. Senders streamed in 16-row chunks.
// edge_part tile = split-bf16 WMMA (hiA*hiB + hiA*loB + loA*hiB), f32 acc.
// W_edge splits stored K-pair-packed in LDS (b32 fragment reads, no repack).
// Gate = (u-1)*rcp((u+1)*(1+f)) with u,f from two multiplies (factored exp).
// ---------------------------------------------------------------------------
__global__ void __launch_bounds__(256) msg_kernel(
    const float* __restrict__ edge_features,
    const unsigned char* __restrict__ mask,
    const float* __restrict__ W_edge,
    const float2* __restrict__ rr_ws, const float2* __restrict__ ss_ws,
    float* __restrict__ out)
{
    // K-pair-packed splits: index [kp * HM + hm], kp = k>>1,
    // dword = bf16(k+1)<<16 | bf16(k)  (hi and lo arrays).
    __shared__ unsigned whi[(En / 2) * HM];   // 16 KB
    __shared__ unsigned wlo[(En / 2) * HM];   // 16 KB
    __shared__ float er2[HM];   // exp(2*recv') for this j
    __shared__ float frj[HM];   // exp(-rn')    for this j
    __shared__ float updbuf[HM];

    const int bj   = blockIdx.x;
    const int b    = bj / Nn;
    const int j    = bj % Nn;
    const int tid  = threadIdx.x;
    const int wave = tid >> 5;
    const int lane = tid & 31;
    const int ln16 = lane & 15;
    const int hiHalf = (lane >= 16) ? 1 : 0;

    // Stage W_edge split into packed hi/lo bf16 pairs in LDS.
    for (int idx = tid; idx < (En / 2) * HM; idx += 256) {
        const int kp = idx / HM, hm = idx % HM;
        float w0 = W_edge[(size_t)(2 * kp) * HM + hm];
        float w1 = W_edge[(size_t)(2 * kp + 1) * HM + hm];
        unsigned hp = bfpack(w1, w0);
        float r0 = w0 - bits_f(fbits(w0) & 0xffff0000u);
        float r1 = w1 - bits_f(fbits(w1) & 0xffff0000u);
        whi[idx] = hp;
        wlo[idx] = bfpack(r1, r0);
    }
    {
        float2 rr = rr_ws[(size_t)bj * HM + tid];
        er2[tid] = rr.x;
        frj[tid] = rr.y;
    }
    __syncthreads();

    const float maskJ = mask[b * Nn + j] ? 1.0f : 0.0f;

    // B fragments for this wave's two hm tiles (invariant over senders).
    // B(32x16) layout: col = lane%16, elems 0..7 -> K = 8*(lane>=16) + e ; elems 8..15 -> K = 16+8*(lane>=16)+(e-8).
    // Element pair (2p,2p+1) == packed dword kp = 4*hiHalf + p + (p>=4 ? 4 : 0).
    v8u bh_p[2], bl_p[2];
    #pragma unroll
    for (int t = 0; t < 2; ++t) {
        const int hmcol = (wave * 2 + t) * 16 + ln16;
        #pragma unroll
        for (int p = 0; p < 8; ++p) {
            const int kp = hiHalf * 4 + p + (p >= 4 ? 4 : 0);
            bh_p[t][p] = whi[kp * HM + hmcol];
            bl_p[t][p] = wlo[kp * HM + hmcol];
        }
    }
    const int hmc0 = (wave * 2 + 0) * 16 + ln16;
    const float er2_0 = er2[hmc0],      fr_0 = frj[hmc0];
    const float er2_1 = er2[hmc0 + 16], fr_1 = frj[hmc0 + 16];

    float upd[2] = {0.0f, 0.0f};
    const int K0 = hiHalf ? 8 : 0;  // A layout K base for this half-wave

    for (int c = 0; c < 4; ++c) {               // sender chunks of 16
        // ---- Gate pair loads: one base pointer, constant immediate offsets.
        // C/D layout: element r -> sender row i = 16c + r + 8*hiHalf.
        const float2* pss = ss_ws +
            ((size_t)b * Nn + c * 16 + hiHalf * 8) * HM + hmc0;
        float2 g0[8], g1[8];
        #pragma unroll
        for (int r = 0; r < 8; ++r) {
            g0[r] = pss[r * HM];
            g1[r] = pss[r * HM + 16];
        }

        // ---- A fragment: row iA = 16c + lane%16.
        const int iA = c * 16 + ln16;
        const float mI = mask[b * Nn + iA] ? maskJ : 0.0f;
        const float* ep = edge_features + (((size_t)b * Nn + iA) * Nn + j) * En;
        // A(16x32) layout: elems 0..7 -> K = K0+e ; elems 8..15 -> K = K0+16+(e-8)
        float4 f0 = *(const float4*)(ep + K0);
        float4 f1 = *(const float4*)(ep + K0 + 4);
        float4 f2 = *(const float4*)(ep + K0 + 16);
        float4 f3 = *(const float4*)(ep + K0 + 20);
        float av[16] = { f0.x, f0.y, f0.z, f0.w,  f1.x, f1.y, f1.z, f1.w,
                         f2.x, f2.y, f2.z, f2.w,  f3.x, f3.y, f3.z, f3.w };
        v8u ah_p, al_p;
        #pragma unroll
        for (int p = 0; p < 8; ++p) {
            float v0 = av[2 * p]     * mI;      // fold dm into ef (dm^2 == dm)
            float v1 = av[2 * p + 1] * mI;
            ah_p[p] = bfpack(v1, v0);
            float r0 = v0 - bits_f(fbits(v0) & 0xffff0000u);
            float r1 = v1 - bits_f(fbits(v1) & 0xffff0000u);
            al_p[p] = bfpack(r1, r0);
        }
        v16bf ah = __builtin_bit_cast(v16bf, ah_p);
        v16bf al = __builtin_bit_cast(v16bf, al_p);

        #pragma unroll
        for (int t = 0; t < 2; ++t) {
            v16bf bh = __builtin_bit_cast(v16bf, bh_p[t]);
            v16bf bl = __builtin_bit_cast(v16bf, bl_p[t]);
            v8f acc = {};
            // split-precision fp32-from-bf16: hi*hi + hi*lo + lo*hi
            acc = __builtin_amdgcn_wmma_f32_16x16x32_bf16(false, ah, false, bh,
                                                          (short)0, acc, false, false);
            acc = __builtin_amdgcn_wmma_f32_16x16x32_bf16(false, ah, false, bl,
                                                          (short)0, acc, false, false);
            acc = __builtin_amdgcn_wmma_f32_16x16x32_bf16(false, al, false, bh,
                                                          (short)0, acc, false, false);
            const float er2v = t ? er2_1 : er2_0;
            const float frv  = t ? fr_1  : fr_0;
            float partial = 0.0f;
            #pragma unroll
            for (int r = 0; r < 8; ++r) {
                const float2 gp = t ? g1[r] : g0[r];
                // tanh*sigmoid via factored exponentials, single v_rcp:
                float u = fminf(er2v * gp.x, 1.0e30f);   // e^{2(recv'+send')}
                float f = frv * gp.y;                    // e^{-(rn'+sn')}
                float g = (u - 1.0f) *
                          __builtin_amdgcn_rcpf((u + 1.0f) * (1.0f + f));
                partial = fmaf(g, acc[r], partial);
            }
            upd[t] += partial;
        }
    }

    // Combine the two row-halves (lanes l and l+16 hold same column).
    #pragma unroll
    for (int t = 0; t < 2; ++t) {
        float tot = upd[t] + __shfl_xor(upd[t], 16);
        if (!hiHalf) updbuf[(wave * 2 + t) * 16 + ln16] = tot;
    }
    __syncthreads();

    // Final LN over M=32 per head: wave w == head h, lane == m.
    float val = updbuf[tid];
    float s = val, s2 = val * val;
    for (int off = 16; off; off >>= 1) {
        s  += __shfl_xor(s,  off);
        s2 += __shfl_xor(s2, off);
    }
    float mu  = s  * (1.0f / 32.0f);
    float var = s2 * (1.0f / 32.0f) - mu * mu;
    out[(size_t)bj * HM + tid] = (val - mu) * rsqrtf(var + EPSf);
}

// ---------------------------------------------------------------------------
extern "C" void kernel_launch(void* const* d_in, const int* in_sizes, int n_in,
                              void* d_out, int out_size, void* d_ws, size_t ws_size,
                              hipStream_t stream) {
    (void)in_sizes; (void)n_in; (void)out_size; (void)ws_size;
    const float*         embeddings    = (const float*)d_in[0];
    const float*         edge_features = (const float*)d_in[1];
    const unsigned char* mask          = (const unsigned char*)d_in[2];
    const float*         node_features = (const float*)d_in[3];
    const float*         W_edge        = (const float*)d_in[4];
    const float*         W_recv        = (const float*)d_in[5];
    const float*         W_send        = (const float*)d_in[6];
    const float*         W_recv_node   = (const float*)d_in[7];
    const float*         W_send_node   = (const float*)d_in[8];
    float* out = (float*)d_out;

    const size_t R = (size_t)Bn * Nn * HM;   // 524288 pairs per buffer
    float2* rr_ws = (float2*)d_ws;           // {e^{2recv'}, e^{-rn'}} : 4 MB
    float2* ss_ws = rr_ws + R;               // {e^{2send'}, e^{-sn'}} : 4 MB

    proj_kernel<<<Bn * Nn, 256, 0, stream>>>(embeddings, mask, node_features,
                                             W_recv, W_send, W_recv_node, W_send_node,
                                             rr_ws, ss_ws);
    msg_kernel<<<Bn * Nn, 256, 0, stream>>>(edge_features, mask, W_edge,
                                            rr_ws, ss_ws, out);
}
